// GRU_17480516895031
// MI455X (gfx1250) — matrix operand order repaired and checked by reading, not compiled, not run on hardware
//
#include <hip/hip_runtime.h>
#include <cstdint>

typedef __bf16 bf16;
typedef __attribute__((ext_vector_type(4)))  float  v4f;
typedef __attribute__((ext_vector_type(8)))  float  v8f;
typedef __attribute__((ext_vector_type(4)))  __bf16 v4bf;
typedef __attribute__((ext_vector_type(8)))  __bf16 v8bf;
typedef __attribute__((ext_vector_type(16))) __bf16 v16bf;

#define LN_EPS 1e-5f

#if __has_builtin(__builtin_amdgcn_global_load_async_to_lds_b64) && \
    __has_builtin(__builtin_amdgcn_s_wait_asynccnt)
#define HAVE_ASYNC_LDS 1
typedef int v2i __attribute__((vector_size(8)));                 // b64 payload type
typedef __attribute__((address_space(1))) v2i* as1_v2i_p;        // global (AS1)
typedef __attribute__((address_space(3))) v2i* as3_v2i_p;        // LDS (AS3)
#endif

// ---------------- helpers ----------------

__global__ void k_cast_bf16(const float* __restrict__ s, bf16* __restrict__ d, int n) {
  int i = (blockIdx.x * blockDim.x + threadIdx.x) * 4;
  if (i < n) {
    v4f v = *(const v4f*)(s + i);
    v4bf o;
    o.x = (bf16)v.x; o.y = (bf16)v.y; o.z = (bf16)v.z; o.w = (bf16)v.w;
    *(v4bf*)(d + i) = o;
  }
}

__global__ void k_zero_bf16(bf16* __restrict__ d, int n) {
  int i = blockIdx.x * blockDim.x + threadIdx.x;
  if (i < n) d[i] = (bf16)0.0f;
}

// inputs [B,T,I] f32 -> seq [T,B,I] bf16 (time-major for the recurrence)
__global__ void k_cast_tr_seq(const float* __restrict__ s, bf16* __restrict__ d,
                              int B, int T, int I) {
  int tb = blockIdx.x;          // tb = t*B + b (output row)
  int t = tb / B, b = tb % B;
  int i = threadIdx.x * 4;      // 128 threads * 4 = 512 = I
  v4f v = *(const v4f*)(s + ((size_t)b * T + t) * I + i);
  v4bf o;
  o.x = (bf16)v.x; o.y = (bf16)v.y; o.z = (bf16)v.z; o.w = (bf16)v.w;
  *(v4bf*)(d + (size_t)tb * I + i) = o;
}

// 512x512 bf16 transpose; load phase goes memory->LDS via the gfx1250 async
// copy path (ASYNCcnt) when available. Tile padded to 36 cols: keeps every
// 4-element chunk 8B-aligned for b64 and breaks bank conflicts on the
// transposed read.
__global__ __launch_bounds__(256)
void k_transpose_bf16(const bf16* __restrict__ src, bf16* __restrict__ dst, int n) {
  __shared__ bf16 tile[32][36];
  int nb = n / 32;
  int bx = blockIdx.x % nb, by = blockIdx.x / nb;
  int r  = threadIdx.x >> 3;        // 0..31
  int c4 = (threadIdx.x & 7) * 4;   // 0,4,...,28
  const bf16* gp = src + ((size_t)(by * 32 + r)) * n + bx * 32 + c4;
  bf16* lp = &tile[r][c4];
#if defined(HAVE_ASYNC_LDS)
  __builtin_amdgcn_global_load_async_to_lds_b64(
      (as1_v2i_p)(uintptr_t)gp, (as3_v2i_p)(uint32_t)(uintptr_t)lp, 0, 0);
  __builtin_amdgcn_s_wait_asynccnt(0);
#else
  *(v4bf*)lp = *(const v4bf*)gp;
#endif
  __syncthreads();
  v4bf o;
  o.x = tile[c4 + 0][r]; o.y = tile[c4 + 1][r];
  o.z = tile[c4 + 2][r]; o.w = tile[c4 + 3][r];
  *(v4bf*)(dst + ((size_t)(bx * 32 + r)) * n + by * 32 + c4) = o;
}

// ---------------- WMMA GEMM: C[M,N] = A[M,K] * W[N,K]^T (+bias)(+addend) ----------------
// K fixed at 512 (I == H == 512 for every matmul in this net). 256 threads =
// 8 waves arranged 2(m) x 2(n) x 2(k-split); each wave owns a 32x32 tile
// (2x2 of v_wmma_f32_16x16x32_bf16) over half of K. Wave-uniform control is
// pinned to SGPRs via readfirstlane so the k-loop is scalar (no EXEC masking
// around WMMA) and unrolls cleanly. k-group 1 dumps partials to LDS; k-group
// 0 reduces and runs the epilogue. M%64==0, N%64==0 for all shapes here.
#define GEMM_K 512

__global__ __launch_bounds__(256)
void k_gemm_bf16(const bf16* __restrict__ A, int lda,
                 const bf16* __restrict__ W, int ldw,
                 const float* __restrict__ bias,
                 const float* __restrict__ addend, int ldadd,
                 float* __restrict__ Cf, long long ldcf, int rowmapB, int rowmapT,
                 bf16* __restrict__ Cbf, int ldcb,
                 int M, int N) {
  __shared__ float redbuf[4][32][36];  // [tile-wave][lane][32 used, pad 36]
  const int lane = threadIdx.x & 31;
  const int wave = __builtin_amdgcn_readfirstlane(threadIdx.x >> 5);  // SGPR
  const int wk   = wave >> 2;        // k-split group 0/1 (scalar)
  const int widx = wave & 3;         // tile index within block (scalar)
  const int wm = widx & 1, wn = widx >> 1;
  const int m0 = blockIdx.y * 64 + wm * 32;
  const int n0 = blockIdx.x * 64 + wn * 32;
  const int half = lane >> 4, l16 = lane & 15;
  (void)M; (void)N;

  v8f acc[2][2] = {};

  const bf16* Ar0 = A + (size_t)(m0 + l16)      * lda;
  const bf16* Ar1 = A + (size_t)(m0 + 16 + l16) * lda;
  const bf16* Wr0 = W + (size_t)(n0 + l16)      * ldw;
  const bf16* Wr1 = W + (size_t)(n0 + 16 + l16) * ldw;

  const int kbeg = wk * (GEMM_K / 2);            // scalar loop bounds
  #pragma unroll 4
  for (int kk = 0; kk < GEMM_K / 2; kk += 32) {
    const int k0 = kbeg + kk;
    const int ka = k0 + half * 8;   // A frag: K 0..7 / 8..15 by lane-half, +16 for hi
    const int kb = k0 + half * 8;   // B frag: K 0..7 / 8..15 by lane-half, +16 for hi
    __builtin_prefetch((const void*)(Ar0 + ka + 64), 0, 1);
    __builtin_prefetch((const void*)(Wr0 + kb + 64), 0, 1);
    v8bf a0lo = *(const v8bf*)(Ar0 + ka);
    v8bf a0hi = *(const v8bf*)(Ar0 + ka + 16);
    v8bf a1lo = *(const v8bf*)(Ar1 + ka);
    v8bf a1hi = *(const v8bf*)(Ar1 + ka + 16);
    v8bf b0lo = *(const v8bf*)(Wr0 + kb);
    v8bf b0hi = *(const v8bf*)(Wr0 + kb + 16);
    v8bf b1lo = *(const v8bf*)(Wr1 + kb);
    v8bf b1hi = *(const v8bf*)(Wr1 + kb + 16);
    v16bf a0 = __builtin_shufflevector(a0lo, a0hi, 0,1,2,3,4,5,6,7,8,9,10,11,12,13,14,15);
    v16bf a1 = __builtin_shufflevector(a1lo, a1hi, 0,1,2,3,4,5,6,7,8,9,10,11,12,13,14,15);
    v16bf b0 = __builtin_shufflevector(b0lo, b0hi, 0,1,2,3,4,5,6,7,8,9,10,11,12,13,14,15);
    v16bf b1 = __builtin_shufflevector(b1lo, b1hi, 0,1,2,3,4,5,6,7,8,9,10,11,12,13,14,15);
    acc[0][0] = __builtin_amdgcn_wmma_f32_16x16x32_bf16(false, a0, false, b0, (short)0, acc[0][0], false, false);
    acc[0][1] = __builtin_amdgcn_wmma_f32_16x16x32_bf16(false, a0, false, b1, (short)0, acc[0][1], false, false);
    acc[1][0] = __builtin_amdgcn_wmma_f32_16x16x32_bf16(false, a1, false, b0, (short)0, acc[1][0], false, false);
    acc[1][1] = __builtin_amdgcn_wmma_f32_16x16x32_bf16(false, a1, false, b1, (short)0, acc[1][1], false, false);
  }

  // cross-k reduction through LDS (wk is scalar -> uniform branch, EXEC untouched)
  if (wk == 1) {
    float* dp = &redbuf[widx][lane][0];
    #pragma unroll
    for (int i = 0; i < 2; ++i)
      #pragma unroll
      for (int j = 0; j < 2; ++j)
        #pragma unroll
        for (int r = 0; r < 8; ++r)
          dp[(i * 2 + j) * 8 + r] = acc[i][j][r];
  }
  __syncthreads();
  if (wk != 0) return;
  {
    const float* sp = &redbuf[widx][lane][0];
    #pragma unroll
    for (int i = 0; i < 2; ++i)
      #pragma unroll
      for (int j = 0; j < 2; ++j)
        #pragma unroll
        for (int r = 0; r < 8; ++r)
          acc[i][j][r] += sp[(i * 2 + j) * 8 + r];
  }

  // epilogue: C/D layout -> lane covers n = n0+j*16+l16, m = m0+i*16+half*8+r
  #pragma unroll
  for (int i = 0; i < 2; ++i) {
    #pragma unroll
    for (int j = 0; j < 2; ++j) {
      const int n = n0 + j * 16 + l16;
      const float bn = bias ? bias[n] : 0.0f;
      #pragma unroll
      for (int r = 0; r < 8; ++r) {
        const int m = m0 + i * 16 + half * 8 + r;
        float v = acc[i][j][r] + bn;
        if (addend) v += addend[(size_t)m * ldadd + n];
        if (Cf) {
          long long row = rowmapB ? (long long)(m % rowmapB) * rowmapT + (m / rowmapB)
                                  : (long long)m;
          Cf[row * ldcf + n] = v;
        }
        if (Cbf) Cbf[(size_t)m * ldcb + n] = (bf16)v;
      }
    }
  }
}

// ---------------- row LayerNorm + activation (one wave per 512-wide row) ----------------
// mode 0: sigmoid -> out1 ; mode 1: sigmoid -> out1(z), out2(1-z) ; mode 2: tanh -> out1
__global__ __launch_bounds__(256)
void k_ln_act(const float* __restrict__ src, int lds,
              const float* __restrict__ gamma, const float* __restrict__ beta,
              int mode, bf16* __restrict__ out1, bf16* __restrict__ out2, int rows) {
  const int wave = threadIdx.x >> 5, lane = threadIdx.x & 31;
  const int row = blockIdx.x * 8 + wave;
  if (row >= rows) return;
  const float* p = src + (size_t)row * lds + lane * 16;
  float xs[16];
  #pragma unroll
  for (int q = 0; q < 4; ++q) {
    v4f v = *(const v4f*)(p + q * 4);
    xs[q * 4 + 0] = v.x; xs[q * 4 + 1] = v.y; xs[q * 4 + 2] = v.z; xs[q * 4 + 3] = v.w;
  }
  float s = 0.0f, s2 = 0.0f;
  #pragma unroll
  for (int t = 0; t < 16; ++t) { s += xs[t]; s2 += xs[t] * xs[t]; }
  for (int o = 16; o > 0; o >>= 1) {
    s  += __shfl_xor(s,  o, 32);
    s2 += __shfl_xor(s2, o, 32);
  }
  const float mean = s * (1.0f / 512.0f);
  const float var  = s2 * (1.0f / 512.0f) - mean * mean;
  const float rstd = rsqrtf(var + LN_EPS);

  float a1[16], a2[16];
  #pragma unroll
  for (int t = 0; t < 16; ++t) {
    const int c = lane * 16 + t;
    const float u = (xs[t] - mean) * rstd * gamma[c] + beta[c];
    if (mode == 2) {
      a1[t] = tanhf(u); a2[t] = 0.0f;
    } else {
      const float sg = 1.0f / (1.0f + __expf(-u));
      a1[t] = sg; a2[t] = 1.0f - sg;
    }
  }
  v8bf o0, o1;
  #pragma unroll
  for (int t = 0; t < 8; ++t) { o0[t] = (bf16)a1[t]; o1[t] = (bf16)a1[t + 8]; }
  bf16* d1 = out1 + (size_t)row * 512 + lane * 16;
  *(v8bf*)d1 = o0; *(v8bf*)(d1 + 8) = o1;
  if (mode == 1) {
    #pragma unroll
    for (int t = 0; t < 8; ++t) { o0[t] = (bf16)a2[t]; o1[t] = (bf16)a2[t + 8]; }
    bf16* d2 = out2 + (size_t)row * 512 + lane * 16;
    *(v8bf*)d2 = o0; *(v8bf*)(d2 + 8) = o1;
  }
}

// ---------------- host orchestration ----------------

extern "C" void kernel_launch(void* const* d_in, const int* in_sizes, int n_in,
                              void* d_out, int out_size, void* d_ws, size_t ws_size,
                              hipStream_t stream) {
  (void)in_sizes; (void)n_in; (void)out_size; (void)ws_size;
  const int B = 512, T = 128, I = 512, H = 512, O = 512, L = 2;

  const float* in_x  = (const float*)d_in[0];
  const float* Wh    = (const float*)d_in[1];  // [L,3,H,H]
  const float* Wx    = (const float*)d_in[2];  // [L,3,H,I]
  const float* bx    = (const float*)d_in[3];  // [L,3,H]
  const float* gamma = (const float*)d_in[4];  // [L,3,H]
  const float* beta  = (const float*)d_in[5];  // [L,3,H]
  const float* Wy    = (const float*)d_in[6];  // [O,H]
  const float* by    = (const float*)d_in[7];  // [O]

  float* outx = (float*)d_out;                     // [B,T,H]
  float* outy = outx + (size_t)B * T * H;          // [B,T,O]

  // ---- workspace carve ----
  char* wsb = (char*)d_ws;
  size_t off = 0;
  auto carve = [&](size_t bytes) -> char* {
    char* p = wsb + off;
    off += (bytes + 255) & ~(size_t)255;
    return p;
  };
  const size_t nWh = (size_t)L * 3 * H * H;
  const size_t nWx = (size_t)L * 3 * H * I;
  const size_t nWy = (size_t)O * H;
  bf16* WhB   = (bf16*)carve(nWh * 2);
  bf16* WxB   = (bf16*)carve(nWx * 2);
  bf16* WyB   = (bf16*)carve(nWy * 2);
  bf16* seq   = (bf16*)carve((size_t)T * B * H * 2);  // x_t / h_t sequence (bf16)
  bf16* hzero = (bf16*)carve((size_t)B * H * 2);
  float* Prz  = (float*)carve((size_t)B * 2 * H * 4);
  float* Pg   = (float*)carve((size_t)B * H * 4);
  float* S    = (float*)carve((size_t)B * H * 4);
  bf16* r_bf  = (bf16*)carve((size_t)B * H * 2);
  bf16* zb_bf = (bf16*)carve((size_t)B * H * 2);
  bf16* om_bf = (bf16*)carve((size_t)B * H * 2);  // 1 - z
  bf16* t1_bf = (bf16*)carve((size_t)B * H * 2);  // r @ h
  bf16* g_bf  = (bf16*)carve((size_t)B * H * 2);
  bf16* hT_bf = (bf16*)carve((size_t)B * H * 2);
  bf16* gT_bf = (bf16*)carve((size_t)B * H * 2);
  bf16* zT_bf = (bf16*)carve((size_t)B * H * 2);

  auto cast = [&](const float* s, bf16* d, size_t n) {
    k_cast_bf16<<<dim3((unsigned)((n / 4 + 255) / 256)), 256, 0, stream>>>(s, d, (int)n);
  };
  auto gemm = [&](const bf16* A, int lda, const bf16* W, int ldw, const float* bias,
                  const float* add, int ldadd, float* Cf, long long ldcf, int rmB, int rmT,
                  bf16* Cbf, int ldcb, int M, int N) {
    dim3 g((unsigned)(N / 64), (unsigned)(M / 64));
    k_gemm_bf16<<<g, 256, 0, stream>>>(A, lda, W, ldw, bias, add, ldadd,
                                       Cf, ldcf, rmB, rmT, Cbf, ldcb, M, N);
  };
  auto ln = [&](const float* src, int lds, const float* gm, const float* bt,
                int mode, bf16* o1, bf16* o2) {
    k_ln_act<<<dim3(B / 8), 256, 0, stream>>>(src, lds, gm, bt, mode, o1, o2, B);
  };
  auto tr = [&](const bf16* s, bf16* d) {
    k_transpose_bf16<<<dim3((H / 32) * (H / 32)), 256, 0, stream>>>(s, d, H);
  };

  // weights -> bf16; inputs -> time-major bf16; h0 = 0
  cast(Wh, WhB, nWh);
  cast(Wx, WxB, nWx);
  cast(Wy, WyB, nWy);
  k_cast_tr_seq<<<dim3((unsigned)(T * B)), 128, 0, stream>>>(in_x, seq, B, T, I);
  k_zero_bf16<<<dim3((B * H + 255) / 256), 256, 0, stream>>>(hzero, B * H);

  for (int l = 0; l < L; ++l) {
    const bf16* Wh_l = WhB + (size_t)l * 3 * H * H;
    const bf16* Wx_l = WxB + (size_t)l * 3 * H * I;
    const float* bx_l = bx    + (size_t)l * 3 * H;
    const float* gm_l = gamma + (size_t)l * 3 * H;
    const float* bt_l = beta  + (size_t)l * 3 * H;
    for (int t = 0; t < T; ++t) {
      const bf16* xt    = seq + (size_t)t * B * I;
      const bf16* hprev = (t == 0) ? hzero : seq + (size_t)(t - 1) * B * H;

      // r,z pre-activations: Prz = x_t @ Wx_rz^T + bx_rz ; Prz += h @ Wh_rz^T
      gemm(xt, I, Wx_l, I, bx_l, nullptr, 0, Prz, 2 * H, 0, 0, nullptr, 0, B, 2 * H);
      gemm(hprev, H, Wh_l, H, nullptr, Prz, 2 * H, Prz, 2 * H, 0, 0, nullptr, 0, B, 2 * H);
      ln(Prz,     2 * H, gm_l,     bt_l,     0, r_bf, nullptr);          // r = sigmoid(LN)
      ln(Prz + H, 2 * H, gm_l + H, bt_l + H, 1, zb_bf, om_bf);           // z, 1-z

      // t1 = r @ h  (via W = h^T so W^T = h)
      tr(hprev, hT_bf);
      gemm(r_bf, H, hT_bf, H, nullptr, nullptr, 0, nullptr, 0, 0, 0, t1_bf, H, B, H);

      // g pre-activation: Pg = x_t @ Wx_g^T + bx_g ; Pg += t1 @ Wh_g^T
      gemm(xt, I, Wx_l + (size_t)2 * H * I, I, bx_l + 2 * H,
           nullptr, 0, Pg, H, 0, 0, nullptr, 0, B, H);
      gemm(t1_bf, H, Wh_l + (size_t)2 * H * H, H, nullptr,
           Pg, H, Pg, H, 0, 0, nullptr, 0, B, H);
      ln(Pg, H, gm_l + 2 * H, bt_l + 2 * H, 2, g_bf, nullptr);           // g = tanh(LN)

      // h_new = (1-z) @ g + h @ z
      tr(g_bf, gT_bf);
      tr(zb_bf, zT_bf);
      gemm(om_bf, H, gT_bf, H, nullptr, nullptr, 0, S, H, 0, 0, nullptr, 0, B, H);
      float* cf = (l == L - 1) ? (outx + (size_t)t * H) : nullptr;       // x[b][t][:]
      gemm(hprev, H, zT_bf, H, nullptr, S, H, cf, (long long)T * H, 0, 0,
           seq + (size_t)t * B * H, H, B, H);                            // also h_t (bf16)
    }
  }

  // y[b,t,:] = x @ Wy^T + by  (A rows are (t,b); swizzle rows to b*T+t on store)
  gemm(seq, H, WyB, H, by, nullptr, 0, outy, O, B, T, nullptr, 0, T * B, O);
}
